// NeighborAttn_40381282517301
// MI455X (gfx1250) — matrix operand order repaired and checked by reading, not compiled, not run on hardware
//
#include <hip/hip_runtime.h>
#include <hip/hip_bf16.h>
#include <math.h>

typedef __bf16 bf16_t;
typedef __attribute__((ext_vector_type(16))) __bf16 v16bf;
typedef __attribute__((ext_vector_type(8)))  __bf16 v8bf;
typedef __attribute__((ext_vector_type(8)))  float  v8f;

#define DIMC 256
#define HH   128
#define WW   128
#define NTOK 64
#define STRX 264   // bf16 row stride for 64x256 tiles (pad 8)
#define STRV 72    // bf16 row stride for [256][64] / [64][64] tiles (pad 8)
#define STRS 68    // f32 row stride for 64x64 scores (pad 4)
#define NTHREADS 512   // 16 waves -> 4 waves/SIMD on one WGP

// ---- fragment loaders (CDNA5 WMMA bf16 16x16x32 layouts, ISA 7.12.2) ----

// A-matrix 16x32: lane%16 = row, lane/16 selects K-half; this code packs K[h*16 .. h*16+15] (contiguous, not ISA table order)
// -> one 32B contiguous chunk per lane (two 16B loads).
static __device__ __forceinline__ v16bf load_afrag(const bf16_t* base, int stride,
                                                   int row0, int k0, int lane) {
  int r = row0 + (lane & 15);
  int hv = lane >> 4;
  const bf16_t* p = base + r * stride + k0 + hv * 16;
  union { v16bf v; v8bf h2[2]; } u;
  u.h2[0] = *(const v8bf*)(p);
  u.h2[1] = *(const v8bf*)(p + 8);
  return u.v;
}

// B-matrix 32x16: lane%16 = col n, lane/16 selects K-half; this code packs K[h*16 .. h*16+15] (contiguous, not ISA table order)
// -> single 32B contiguous chunk per lane when the K-major matrix is stored
// row-major as [n][k] (e.g. K^T from row-major K, or V^T stored as [c][m]).
static __device__ __forceinline__ v16bf load_bfrag_rm(const bf16_t* base, int stride,
                                                      int n0, int k0, int lane) {
  const bf16_t* p = base + (n0 + (lane & 15)) * stride + k0 + (lane >> 4) * 16;
  union { v16bf v; v8bf h2[2]; } u;
  u.h2[0] = *(const v8bf*)(p);
  u.h2[1] = *(const v8bf*)(p + 8);
  return u.v;
}

// Pre-swizzled weights: fragment for (kt,nt,lane) stored contiguously (32B).
static __device__ __forceinline__ v16bf load_bfrag_w(const bf16_t* wbase, int ntiles,
                                                     int kt, int nt, int lane) {
  const bf16_t* p = wbase + ((size_t)(((kt * ntiles + nt) << 5) + lane) << 4);
  union { v16bf v; v8bf h2[2]; } u;
  u.h2[0] = *(const v8bf*)(p);
  u.h2[1] = *(const v8bf*)(p + 8);
  return u.v;
}

#define WMMA_BF16(a, b, c) \
  __builtin_amdgcn_wmma_f32_16x16x32_bf16(false, (a), false, (b), (short)0, (c), false, false)

static __device__ __forceinline__ float gelu_exact(float v) {
  return 0.5f * v * (1.0f + erff(v * 0.70710678118654752f));
}

// ---------------- weight pre-swizzle: f32 KxN row-major -> B-fragment order ----------------
__global__ __launch_bounds__(256) void prep_weights_kernel(
    const float* __restrict__ qkv_w, const float* __restrict__ z_w,
    const float* __restrict__ proj_w,
    bf16_t* __restrict__ wq, bf16_t* __restrict__ wz, bf16_t* __restrict__ wp) {
  int fid = blockIdx.x * 256 + threadIdx.x;   // lane-fragment id
  const float* W; bf16_t* O; int N; int f;
  if (fid < 12288)      { W = qkv_w;  O = wq; N = 768; f = fid; }
  else if (fid < 16384) { W = z_w;    O = wz; N = 256; f = fid - 12288; }
  else if (fid < 20480) { W = proj_w; O = wp; N = 256; f = fid - 16384; }
  else return;
  int ntiles = N >> 4;
  int kt   = f / (ntiles * 32);
  int r    = f - kt * (ntiles * 32);
  int nt   = r >> 5;
  int lane = r & 31;
  int hv   = lane >> 4;
  int n    = nt * 16 + (lane & 15);
  bf16_t* o = O + ((size_t)f << 4);
#pragma unroll
  for (int idx = 0; idx < 16; ++idx) {
    int k = kt * 32 + hv * 16 + idx;
    o[idx] = (bf16_t)W[k * N + n];
  }
}

// ---------------- fused per-window kernel ----------------
__global__ __launch_bounds__(NTHREADS) void cswin_window_kernel(
    const float* __restrict__ x,
    const bf16_t* __restrict__ wqkv, const float* __restrict__ qkv_b,
    const bf16_t* __restrict__ wz,   const float* __restrict__ z_b,
    const bf16_t* __restrict__ wproj,const float* __restrict__ proj_b,
    const float* __restrict__ dw_w,  const float* __restrict__ dw_b,
    float* __restrict__ out) {
  extern __shared__ float4 smem4[];
  bf16_t* sX    = (bf16_t*)smem4;            // 64 x STRX  (x tile, later gated attn out)
  bf16_t* sQ    = sX    + NTOK * STRX;       // 64 x STRX
  bf16_t* sK    = sQ    + NTOK * STRX;       // 64 x STRX
  bf16_t* sVt   = sK    + NTOK * STRX;       // 256 x STRV (V transposed: [c][token])
  bf16_t* sLepe = sVt   + DIMC * STRV;       // 256 x STRV
  bf16_t* sP    = sLepe + DIMC * STRV;       // 64 x STRV  (softmax probs, bf16)
  float*  sS    = (float*)(sP + NTOK * STRV);// 64 x STRS  (scores, f32)
  float*  sW9   = sS + NTOK * STRS;          // 256 x 9    (depthwise weights)

  const int tid  = threadIdx.x;
  const int lane = tid & 31;
  const int wv   = tid >> 5;                 // 0..15
  const int hv   = lane >> 4;
  const int l15  = lane & 15;

  const int win = blockIdx.x;                // 2048 windows
  const int bb  = win >> 8;
  const int wr  = win & 255;
  const int h0  = (wr >> 4) * 8;
  const int w0  = (wr & 15) * 8;

  // ---- stage 0: load depthwise weights + x window (64 tok x 256 ch) -> bf16 LDS ----
  for (int i = tid; i < DIMC * 9; i += NTHREADS) sW9[i] = dw_w[i];
  const float* xb = x + (size_t)bb * DIMC * HH * WW;
  for (int s = tid; s < DIMC * 8; s += NTHREADS) {
    int c = s >> 3, r = s & 7;
    const float* p = xb + (size_t)c * (HH * WW) + (size_t)(h0 + r) * WW + w0;
    float4 lo = *(const float4*)p;
    float4 hi = *(const float4*)(p + 4);
    bf16_t* d = sX + (r * 8) * STRX + c;
    d[0 * STRX] = (bf16_t)lo.x; d[1 * STRX] = (bf16_t)lo.y;
    d[2 * STRX] = (bf16_t)lo.z; d[3 * STRX] = (bf16_t)lo.w;
    d[4 * STRX] = (bf16_t)hi.x; d[5 * STRX] = (bf16_t)hi.y;
    d[6 * STRX] = (bf16_t)hi.z; d[7 * STRX] = (bf16_t)hi.w;
  }
  __syncthreads();

  // ---- stage 1: z = gelu(x @ z_w + z_b), kept in registers (gate applied stage 5) ----
  // Tile->wave mapping MUST match stage 5 so the D-fragment (m,c) mapping is identical.
  float zr[4][8];
#pragma unroll
  for (int t = 0; t < 4; ++t) {
    int idx = wv * 4 + t;                 // 0..63
    int mt = idx >> 4, nt = idx & 15;
    v8f acc = {};
#pragma unroll
    for (int kt = 0; kt < 8; ++kt) {
      v16bf a = load_afrag(sX, STRX, mt * 16, kt * 32, lane);
      v16bf b = load_bfrag_w(wz, 16, kt, nt, lane);
      acc = WMMA_BF16(a, b, acc);
    }
    float bias = z_b[nt * 16 + l15];
#pragma unroll
    for (int j = 0; j < 8; ++j) zr[t][j] = gelu_exact(acc[j] + bias);
  }

  // ---- stage 2: qkv = x @ qkv_w + b  (64x256x768); q scaled, v stored transposed ----
  const float qscale = 0.0625f;           // 256^-0.5
  for (int t = 0; t < 12; ++t) {
    int idx = wv + 16 * t;                // 0..191
    int mt = idx & 3, nt = idx >> 2;      // nt 0..47
    v8f acc = {};
#pragma unroll
    for (int kt = 0; kt < 8; ++kt) {
      v16bf a = load_afrag(sX, STRX, mt * 16, kt * 32, lane);
      v16bf b = load_bfrag_w(wqkv, 48, kt, nt, lane);
      acc = WMMA_BF16(a, b, acc);
    }
    int col = nt * 16 + l15;
    float bias = qkv_b[col];
#pragma unroll
    for (int j = 0; j < 8; ++j) {
      int m = mt * 16 + hv * 8 + j;
      float v = acc[j] + bias;
      if (nt < 16)       sQ[m * STRX + col]          = (bf16_t)(v * qscale);
      else if (nt < 32)  sK[m * STRX + (col - 256)]  = (bf16_t)v;
      else               sVt[(col - 512) * STRV + m] = (bf16_t)v;  // [c][token]
    }
  }
  __syncthreads();

  // ---- stage 3a: LePE depthwise 3x3 per window (zero padded) on V ----
  for (int i = tid; i < DIMC * NTOK; i += NTHREADS) {
    int c = i >> 6, tk = i & 63;
    int r = tk >> 3, w = tk & 7;
    const float* wp9 = sW9 + c * 9;
    float s = dw_b[c];
#pragma unroll
    for (int dr = 0; dr < 3; ++dr) {
#pragma unroll
      for (int dc = 0; dc < 3; ++dc) {
        int rr = r + dr - 1, cc = w + dc - 1;
        if (rr >= 0 && rr < 8 && cc >= 0 && cc < 8)
          s += (float)sVt[c * STRV + rr * 8 + cc] * wp9[dr * 3 + dc];
      }
    }
    sLepe[c * STRV + tk] = (bf16_t)s;
  }

  // ---- stage 3b: S = Qs @ K^T (64x64, K=256), one 16x16 tile per wave ----
  {
    int mt = wv >> 2, nt = wv & 3;
    v8f acc = {};
#pragma unroll
    for (int kt = 0; kt < 8; ++kt) {
      v16bf a = load_afrag(sQ, STRX, mt * 16, kt * 32, lane);
      v16bf b = load_bfrag_rm(sK, STRX, nt * 16, kt * 32, lane);  // row-major K == B-frag of K^T
      acc = WMMA_BF16(a, b, acc);
    }
#pragma unroll
    for (int j = 0; j < 8; ++j)
      sS[(mt * 16 + hv * 8 + j) * STRS + nt * 16 + l15] = acc[j];
  }
  __syncthreads();

  // ---- stage 4: softmax rows -> bf16 probs ----
  if (tid < NTOK) {
    float* row = sS + tid * STRS;
    float mx = -1e30f;
    for (int n = 0; n < NTOK; ++n) mx = fmaxf(mx, row[n]);
    float sum = 0.f;
    for (int n = 0; n < NTOK; ++n) { float e = __expf(row[n] - mx); row[n] = e; sum += e; }
    float inv = 1.0f / sum;
    for (int n = 0; n < NTOK; ++n) sP[tid * STRV + n] = (bf16_t)(row[n] * inv);
  }
  __syncthreads();

  // ---- stage 5: O = (P @ V + lepe) * z  -> sX (reused); same tile map as stage 1 ----
#pragma unroll
  for (int t = 0; t < 4; ++t) {
    int idx = wv * 4 + t;                 // 0..63
    int mt = idx >> 4, nt = idx & 15;     // nt = channel tile 0..15
    v8f acc = {};
#pragma unroll
    for (int kt = 0; kt < 2; ++kt) {
      v16bf a = load_afrag(sP, STRV, mt * 16, kt * 32, lane);
      v16bf b = load_bfrag_rm(sVt, STRV, nt * 16, kt * 32, lane); // sVt[c][m] == B-frag of V
      acc = WMMA_BF16(a, b, acc);
    }
    int c = nt * 16 + l15;
#pragma unroll
    for (int j = 0; j < 8; ++j) {
      int m = mt * 16 + hv * 8 + j;
      float v = (acc[j] + (float)sLepe[c * STRV + m]) * zr[t][j];
      sX[m * STRX + c] = (bf16_t)v;
    }
  }
  __syncthreads();

  // ---- stage 6: out = O @ proj_w + proj_b, scatter to NCHW (two float4/lane/tile) ----
#pragma unroll
  for (int t = 0; t < 4; ++t) {
    int idx = wv * 4 + t;                 // 0..63
    int mt = idx >> 4, nt = idx & 15;
    v8f acc = {};
#pragma unroll
    for (int kt = 0; kt < 8; ++kt) {
      v16bf a = load_afrag(sX, STRX, mt * 16, kt * 32, lane);
      v16bf b = load_bfrag_w(wproj, 16, kt, nt, lane);
      acc = WMMA_BF16(a, b, acc);
    }
    int c = nt * 16 + l15;
    float bias = proj_b[c];
    int mbase = mt * 16 + hv * 8;         // multiple of 8 -> one window row, w = j
    float* op = out + ((size_t)bb * DIMC + c) * (HH * WW)
                    + (size_t)(h0 + (mbase >> 3)) * WW + w0;
    float4 o1 = { acc[0] + bias, acc[1] + bias, acc[2] + bias, acc[3] + bias };
    float4 o2 = { acc[4] + bias, acc[5] + bias, acc[6] + bias, acc[7] + bias };
    *(float4*)(op)     = o1;
    *(float4*)(op + 4) = o2;
  }
}

extern "C" void kernel_launch(void* const* d_in, const int* in_sizes, int n_in,
                              void* d_out, int out_size, void* d_ws, size_t ws_size,
                              hipStream_t stream) {
  const float* x      = (const float*)d_in[0];
  const float* qkv_w  = (const float*)d_in[1];
  const float* qkv_b  = (const float*)d_in[2];
  const float* proj_w = (const float*)d_in[3];
  const float* proj_b = (const float*)d_in[4];
  const float* z_w    = (const float*)d_in[5];
  const float* z_b    = (const float*)d_in[6];
  const float* dw_w   = (const float*)d_in[7];
  const float* dw_b   = (const float*)d_in[8];
  float* out = (float*)d_out;

  // ws: pre-swizzled bf16 weights (qkv 384KB, z 128KB, proj 128KB)
  bf16_t* wq = (bf16_t*)d_ws;
  bf16_t* wz = wq + 12288 * 16;
  bf16_t* wp = wz + 4096 * 16;

  prep_weights_kernel<<<80, 256, 0, stream>>>(qkv_w, z_w, proj_w, wq, wz, wp);

  const size_t smem =
      (size_t)3 * NTOK * STRX * 2 +   // sX,sQ,sK
      (size_t)2 * DIMC * STRV * 2 +   // sVt,sLepe
      (size_t)NTOK * STRV * 2 +       // sP
      (size_t)NTOK * STRS * 4 +       // sS
      (size_t)DIMC * 9 * 4;           // sW9   => 210944 B (< 320KB WGP LDS)

  cswin_window_kernel<<<2048, NTHREADS, smem, stream>>>(
      x, wq, qkv_b, wz, z_b, wp, proj_b, dw_w, dw_b, out);
}
